// MyRNN_36077725287162
// MI455X (gfx1250) — matrix operand order repaired and checked by reading, not compiled, not run on hardware
//
#include <hip/hip_runtime.h>
#include <hip/hip_bf16.h>

typedef __attribute__((ext_vector_type(16))) __bf16 v16bf;
typedef __attribute__((ext_vector_type(8)))  float  v8f;

using bf16_t = __hip_bfloat16;

#define T_STEPS 512
#define BATCH   64
#define HID     1024
#define INP     1024
#define NWG_RNN 64
#define SW_PITCH 1032   // padded LDS row pitch in bf16 (516 dwords -> conflict-free b128 frags)

union FragU {
    v16bf bf;
    uint4 u4[2];
};

// A-matrix 16x32 bf16 fragment (M x K), row-major source, ld in elements.
// lane<16: row=lane, K={k..k+15}; lane>=16: row=lane-16, K={k+16..k+31}.
__device__ inline v16bf load_frag_a(const bf16_t* __restrict__ base, int ld) {
    const int lane = threadIdx.x & 31;
    const int row  = lane & 15;
    const int koff = (lane < 16) ? 0 : 16;
    FragU f;
    f.u4[0] = *reinterpret_cast<const uint4*>(base + row * ld + koff);
    f.u4[1] = *reinterpret_cast<const uint4*>(base + row * ld + koff + 8);
    return f.bf;
}

// B-matrix 32x16 bf16 fragment (K x N) with B[k][n] = W[n][k], W row-major (n, k).
// lane<16: n=lane, K={k..k+15}; lane>=16: n=lane-16, K={k+16..k+31}. 32 contiguous bytes.
__device__ inline v16bf load_frag_b(const bf16_t* __restrict__ base, int ld) {
    const int lane = threadIdx.x & 31;
    const int row  = lane & 15;
    const int koff = (lane < 16) ? 0 : 16;
    const uint4* p = reinterpret_cast<const uint4*>(base + row * ld + koff);
    FragU f;
    f.u4[0] = p[0];
    f.u4[1] = p[1];
    return f.bf;
}

// Same B-fragment but from the padded LDS tile (pitch SW_PITCH).
__device__ inline v16bf load_frag_b_lds(const bf16_t* base) {
    const int lane = threadIdx.x & 31;
    const int row  = lane & 15;
    const int koff = (lane < 16) ? 0 : 16;
    const uint4* p = reinterpret_cast<const uint4*>(base + row * SW_PITCH + koff);
    FragU f;
    f.u4[0] = p[0];
    f.u4[1] = p[1];
    return f.bf;
}

// input (B,T,I) f32 -> Xbf[(t*64+b), i] bf16
__global__ void k_cvt_input(const float* __restrict__ in, bf16_t* __restrict__ xbf) {
    const int vid = blockIdx.x * blockDim.x + threadIdx.x;  // over B*T*I/4 = 8388608
    const int i4 = vid & 255;           // I/4 = 256
    const int t  = (vid >> 8) & 511;
    const int b  = vid >> 17;
    const float4 v = reinterpret_cast<const float4*>(in)[((b << 9) + t) * 256 + i4];
    __align__(8) bf16_t tmp[4];
    tmp[0] = __float2bfloat16(v.x);
    tmp[1] = __float2bfloat16(v.y);
    tmp[2] = __float2bfloat16(v.z);
    tmp[3] = __float2bfloat16(v.w);
    reinterpret_cast<uint2*>(xbf)[((t << 6) + b) * 256 + i4] = *reinterpret_cast<const uint2*>(tmp);
}

// generic f32 -> bf16 (exact grid: n/4 threads)
__global__ void k_cvt_w(const float* __restrict__ w, bf16_t* __restrict__ o) {
    const int vid = blockIdx.x * blockDim.x + threadIdx.x;
    const float4 v = reinterpret_cast<const float4*>(w)[vid];
    __align__(8) bf16_t tmp[4];
    tmp[0] = __float2bfloat16(v.x);
    tmp[1] = __float2bfloat16(v.y);
    tmp[2] = __float2bfloat16(v.z);
    tmp[3] = __float2bfloat16(v.w);
    reinterpret_cast<uint2*>(o)[vid] = *reinterpret_cast<const uint2*>(tmp);
}

// bias = bias_ih + bias_hh; zero h0 ping buffer; zero step counters
__global__ void k_prep(const float* __restrict__ bih, const float* __restrict__ bhh,
                       float* __restrict__ bias, uint4* __restrict__ h0,
                       uint4* __restrict__ cnt) {
    const int id = blockIdx.x * 256 + threadIdx.x;
    if (id < 256) {
        const float4 a = reinterpret_cast<const float4*>(bih)[id];
        const float4 b = reinterpret_cast<const float4*>(bhh)[id];
        float4 r;
        r.x = a.x + b.x; r.y = a.y + b.y; r.z = a.z + b.z; r.w = a.w + b.w;
        reinterpret_cast<float4*>(bias)[id] = r;
    } else if (id < 256 + 8192) {            // h0: 64*1024 bf16 = 8192 uint4
        h0[id - 256] = uint4{0u, 0u, 0u, 0u};
    } else if (id < 256 + 8192 + 128) {      // counters: 512 u32 = 128 uint4
        cnt[id - 256 - 8192] = uint4{0u, 0u, 0u, 0u};
    }
}

// x_proj GEMM: out(32768 x 1024) = Xbf * Wih^T + bias, f32 result.
// One wave computes a 64x64 tile (4 M-frags x 4 N-frags = 16 accumulators).
// 8 waves/block share the same 64-row A strip (WGP$ reuse).
__global__ __launch_bounds__(256, 1) void k_xproj(const bf16_t* __restrict__ X,
                                                  const bf16_t* __restrict__ W,
                                                  const float* __restrict__ bias,
                                                  float* __restrict__ out) {
    const int waveId = blockIdx.x * 8 + (threadIdx.x >> 5);
    const int mb = waveId >> 4;   // 0..511 : 64-row strip
    const int nb = waveId & 15;   // 0..15  : 64-col strip
    const int lane = threadIdx.x & 31;

    const bf16_t* Abase = X + (size_t)mb * 64 * INP;
    const bf16_t* Bbase = W + (size_t)nb * 64 * INP;

    v8f acc[4][4];
#pragma unroll
    for (int i = 0; i < 4; ++i)
#pragma unroll
        for (int j = 0; j < 4; ++j)
            acc[i][j] = v8f{};

    for (int k0 = 0; k0 < INP; k0 += 32) {
        v16bf a[4], b[4];
#pragma unroll
        for (int i = 0; i < 4; ++i) a[i] = load_frag_a(Abase + i * 16 * INP + k0, INP);
#pragma unroll
        for (int j = 0; j < 4; ++j) b[j] = load_frag_b(Bbase + j * 16 * INP + k0, INP);
#pragma unroll
        for (int i = 0; i < 4; ++i)
#pragma unroll
            for (int j = 0; j < 4; ++j)
                acc[i][j] = __builtin_amdgcn_wmma_f32_16x16x32_bf16(
                    false, a[i], false, b[j], (short)0, acc[i][j], false, false);
    }

    const int nlo = lane & 15;
    const int mhi = (lane < 16) ? 0 : 8;
#pragma unroll
    for (int j = 0; j < 4; ++j) {
        const int col = nb * 64 + j * 16 + nlo;
        const float bv = bias[col];
#pragma unroll
        for (int i = 0; i < 4; ++i)
#pragma unroll
            for (int r = 0; r < 8; ++r) {
                const int row = mb * 64 + i * 16 + mhi + r;
                out[(size_t)row * HID + col] = acc[i][j][r] + bv;
            }
    }
}

// Persistent recurrence: 64 WGs x 128 threads. WG g owns output cols [16g,16g+16),
// wave w owns batch rows [16w,16w+16). Whh slice lives in LDS (immune to the
// per-step global_inv), preloaded via async global->LDS copies.
__global__ __launch_bounds__(128, 1) void k_rnn(const bf16_t* __restrict__ Whh,
                                                float* __restrict__ out,
                                                bf16_t* __restrict__ hA,
                                                bf16_t* __restrict__ hB,
                                                float* __restrict__ hlast,
                                                unsigned* __restrict__ counters) {
    __shared__ bf16_t sW[16 * SW_PITCH];   // 16 rows x 1032 bf16 = 33 KB

    const int g = blockIdx.x;            // 0..63 column block
    const int tid = threadIdx.x;
    const int w = tid >> 5;              // 0..3 row block
    const int lane = tid & 31;
    const int n0 = g * 16;
    const int m0 = w * 16;
    const int nlo = lane & 15;
    const int mhi = (lane < 16) ? 0 : 8;

    // Async preload Whh[n0..n0+15][:] -> LDS, 16B per thread per row (128 thr * 16B = one row).
    {
        const bf16_t* gsrc = Whh + (size_t)n0 * HID + tid * 8;
#pragma unroll
        for (int r = 0; r < 16; ++r) {
            const unsigned lds_addr =
                (unsigned)(uintptr_t)(sW + r * SW_PITCH + tid * 8);
            const bf16_t* gp = gsrc + r * HID;
            asm volatile("global_load_async_to_lds_b128 %0, %1, off"
                         :: "v"(lds_addr), "v"(gp) : "memory");
        }
        asm volatile("s_wait_asynccnt 0x0" ::: "memory");
    }
    __syncthreads();

    bf16_t* hbuf[2] = { hA, hB };

    for (int t = 0; t < T_STEPS; ++t) {
        const bf16_t* hc = hbuf[t & 1];
        bf16_t* hn = hbuf[(t + 1) & 1];
        float* outp = out + (size_t)t * BATCH * HID;
        const bf16_t* hrow = hc + m0 * HID;

        // Two partial accumulators break the serial WMMA dependency chain.
        v8f acc0 = v8f{}, acc1 = v8f{};
        for (int k0 = 0; k0 < HID; k0 += 64) {
            const v16bf a0 = load_frag_a(hrow + k0, HID);
            const v16bf b0 = load_frag_b_lds(sW + k0);
            const v16bf a1 = load_frag_a(hrow + k0 + 32, HID);
            const v16bf b1 = load_frag_b_lds(sW + k0 + 32);
            acc0 = __builtin_amdgcn_wmma_f32_16x16x32_bf16(
                false, a0, false, b0, (short)0, acc0, false, false);
            acc1 = __builtin_amdgcn_wmma_f32_16x16x32_bf16(
                false, a1, false, b1, (short)0, acc1, false, false);
        }
        const v8f acc = acc0 + acc1;

#pragma unroll
        for (int r = 0; r < 8; ++r) {
            const int row = m0 + mhi + r;
            const int col = n0 + nlo;
            const size_t idx = (size_t)row * HID + col;
            const float gsum = outp[idx] + acc[r];
            const float y = tanhf(gsum);
            outp[idx] = y;
            hn[idx] = __float2bfloat16(y);
            if (t == T_STEPS - 1) hlast[idx] = y;
        }

        if (t + 1 < T_STEPS) {
            // Release: our h/out stores must reach L2 (device scope) before signaling.
            asm volatile("s_wait_storecnt 0x0" ::: "memory");
            __syncthreads();
            if (tid == 0) {
                __hip_atomic_fetch_add(&counters[t], 1u, __ATOMIC_ACQ_REL,
                                       __HIP_MEMORY_SCOPE_AGENT);
                while (__hip_atomic_load(&counters[t], __ATOMIC_RELAXED,
                                         __HIP_MEMORY_SCOPE_AGENT) < NWG_RNN) {
                    __builtin_amdgcn_s_sleep(1);
                }
            }
            __syncthreads();
            // Acquire: invalidate the (read-only) WGP cache; LDS-resident Whh survives.
            asm volatile("global_inv scope:SCOPE_DEV\n\t"
                         "s_wait_loadcnt 0x0" ::: "memory");
        }
    }
}

extern "C" void kernel_launch(void* const* d_in, const int* in_sizes, int n_in,
                              void* d_out, int out_size, void* d_ws, size_t ws_size,
                              hipStream_t stream) {
    (void)in_sizes; (void)n_in; (void)out_size; (void)ws_size;
    const float* input = (const float*)d_in[0];   // (64, 512, 1024)
    const float* wih   = (const float*)d_in[1];   // (1, 1024, 1024)
    const float* whh   = (const float*)d_in[2];   // (1, 1024, 1024)
    const float* bih   = (const float*)d_in[3];   // (1, 1024)
    const float* bhh   = (const float*)d_in[4];   // (1, 1024)

    float* out   = (float*)d_out;                               // (T,B,H) then (B,H)
    float* hlast = out + (size_t)T_STEPS * BATCH * HID;

    char* ws = (char*)d_ws;
    bf16_t* Xbf   = (bf16_t*)ws;                                 // 64 MB
    bf16_t* Wih   = (bf16_t*)(ws + (size_t)64 * 1024 * 1024);    // 2 MB
    bf16_t* Whh   = Wih + 1024 * 1024;                           // 2 MB
    float*  bias  = (float*)(Whh + 1024 * 1024);                 // 4 KB
    bf16_t* h0    = (bf16_t*)(bias + 1024);                      // 128 KB
    bf16_t* h1    = h0 + BATCH * HID;                            // 128 KB
    unsigned* cnt = (unsigned*)(h1 + BATCH * HID);               // 2 KB

    k_cvt_input<<<32768, 256, 0, stream>>>(input, Xbf);
    k_cvt_w<<<1024, 256, 0, stream>>>(wih, Wih);
    k_cvt_w<<<1024, 256, 0, stream>>>(whh, Whh);
    k_prep<<<34, 256, 0, stream>>>(bih, bhh, bias, (uint4*)h0, (uint4*)cnt);
    k_xproj<<<1024, 256, 0, stream>>>(Xbf, Wih, bias, out);
    k_rnn<<<NWG_RNN, 128, 0, stream>>>(Whh, out, h0, h1, hlast, cnt);
}
